// Discriminator_7241314861144
// MI455X (gfx1250) — compile-verified, matrix operand order checked by reading, not run on hardware
//
#include <hip/hip_runtime.h>

// scores[b] = features[b] . (weight @ summary[b])
// = row-sum( F .* (S @ W^T) ), bf16 WMMA, f32 accumulate.
// W is pre-converted to bf16 in scratch; W chunks are staged to LDS with
// gfx1250 async copies (global_load_async_to_lds_b128, ASYNCcnt-tracked),
// double-buffered so copy latency hides under the WMMA stream.

typedef __bf16 bf16_t;
typedef __attribute__((ext_vector_type(4)))  __bf16 v4bf;
typedef __attribute__((ext_vector_type(8)))  __bf16 v8bf;
typedef __attribute__((ext_vector_type(16))) __bf16 v16bf;
typedef __attribute__((ext_vector_type(8)))  float  v8f;
typedef __attribute__((ext_vector_type(4)))  float  v4f;

#define HID            512
#define KSTEPS         16          // 512 / 32 (K per WMMA)
#define ITILES         32          // 512 / 16 (N tiles)
#define LDSW           520         // padded LDS row stride in bf16 elems
#define BLOCK          256         // 8 waves
#define ROWS_PER_BLOCK 128         // 16 rows per wave
#define CHUNK_ELEMS    (16 * LDSW) // one W chunk in LDS (bf16 elems)
#define CHUNK_BYTES    (CHUNK_ELEMS * 2)

static __device__ __forceinline__ v8bf cvt8(v4f a, v4f b) {
  v8bf r;
  r[0] = (__bf16)a[0]; r[1] = (__bf16)a[1]; r[2] = (__bf16)a[2]; r[3] = (__bf16)a[3];
  r[4] = (__bf16)b[0]; r[5] = (__bf16)b[1]; r[6] = (__bf16)b[2]; r[7] = (__bf16)b[3];
  return r;
}

// ---- one-time W f32 -> bf16 conversion into scratch (1MB -> 512KB, L2-hot)
__global__ __launch_bounds__(256)
void convert_w_bf16(const float* __restrict__ W, bf16_t* __restrict__ Wb) {
  const int idx4 = blockIdx.x * 256 + threadIdx.x;   // each thread: 4 elems
  v4f wv = *(const v4f*)(W + (size_t)idx4 * 4);
  v4bf b;
  b[0] = (__bf16)wv[0]; b[1] = (__bf16)wv[1];
  b[2] = (__bf16)wv[2]; b[3] = (__bf16)wv[3];
  *(v4bf*)(Wb + (size_t)idx4 * 4) = b;
}

__global__ __launch_bounds__(BLOCK, 2)
void bilinear_score_wmma(const float* __restrict__ F,
                         const float* __restrict__ S,
                         const bf16_t* __restrict__ Wb,
                         float* __restrict__ out)
{
  __shared__ __align__(16) unsigned short lds_w_raw[2 * CHUNK_ELEMS];
  bf16_t* lds_w = reinterpret_cast<bf16_t*>(lds_w_raw);

  const int tid  = threadIdx.x;
  const int lane = tid & 31;
  const int wave = tid >> 5;
  const int m    = lane & 15;   // A: row M / B: col N / C: col N
  const int h    = lane >> 4;   // lane-half selects K octet

  const int waveRow = blockIdx.x * ROWS_PER_BLOCK + wave * 16;

  // ---- per-thread async-copy geometry: 4 x b128 granules per chunk
  // LDS flat-pointer low 32 bits == LDS byte offset (ISA aperture rule)
  const unsigned ldsbase = (unsigned)(uintptr_t)(&lds_w_raw[0]);
  const bf16_t* gbase[4];
  unsigned      loff[4];
  #pragma unroll
  for (int v = 0; v < 4; ++v) {
    const int idx8 = (v * BLOCK + tid) * 8;     // bf16 elem index in chunk
    const int row  = idx8 >> 9;                 // 0..15
    const int col  = idx8 & (HID - 1);          // mult of 8 -> 16B aligned
    gbase[v] = Wb + ((size_t)row << 9) + col;
    loff[v]  = ldsbase + (unsigned)(row * LDSW + col) * 2u;
  }

  // ---- kick off async copy of chunk 0 into buffer 0
  #pragma unroll
  for (int v = 0; v < 4; ++v) {
    const bf16_t* gp = gbase[v];
    asm volatile("global_load_async_to_lds_b128 %0, %1, off"
                 :: "v"(loff[v]), "v"(gp) : "memory");
  }

  // ---- A operand: this wave's 16 summary rows, full K=512, in VGPRs
  // 16-bit A layout: elems 0..7 = K[8h..8h+8), elems 8..15 = K[16+8h..+8)
  v16bf afrag[KSTEPS];
  {
    const float* srow = S + (size_t)(waveRow + m) * HID;
    #pragma unroll
    for (int j = 0; j < KSTEPS; ++j) {
      const float* p0 = srow + 32 * j + 8 * h;
      v4f f0 = __builtin_nontemporal_load((const v4f*)(p0));
      v4f f1 = __builtin_nontemporal_load((const v4f*)(p0 + 4));
      v4f f2 = __builtin_nontemporal_load((const v4f*)(p0 + 16));
      v4f f3 = __builtin_nontemporal_load((const v4f*)(p0 + 20));
      afrag[j] = __builtin_shufflevector(cvt8(f0, f1), cvt8(f2, f3),
                     0,1,2,3,4,5,6,7,8,9,10,11,12,13,14,15);
    }
  }

  asm volatile("s_wait_asynccnt 0x0" ::: "memory");
  __syncthreads();   // chunk 0 resident

  float score[8] = {0.f,0.f,0.f,0.f,0.f,0.f,0.f,0.f};

  for (int it = 0; it < ITILES; ++it) {
    const int cur = it & 1;

    // ---- fire async copies for the next chunk into the other buffer
    if (it + 1 < ITILES) {
      const unsigned bufoff = (unsigned)((it + 1) & 1) * CHUNK_BYTES;
      const size_t   goff   = (size_t)(it + 1) * 16 * HID;
      #pragma unroll
      for (int v = 0; v < 4; ++v) {
        const bf16_t* gp = gbase[v] + goff;
        const unsigned lo = loff[v] + bufoff;
        asm volatile("global_load_async_to_lds_b128 %0, %1, off"
                     :: "v"(lo), "v"(gp) : "memory");
      }
    }

    // ---- 16 x v_wmma_f32_16x16x32_bf16 over K=512 from current buffer
    const bf16_t* bbase = lds_w + cur * CHUNK_ELEMS + m * LDSW + 8 * h;
    v8f acc = {};
    #pragma unroll
    for (int j = 0; j < KSTEPS; ++j) {
      const bf16_t* bp = bbase + 32 * j;
      v8bf blo = *reinterpret_cast<const v8bf*>(bp);        // K[8h..8h+8)
      v8bf bhi = *reinterpret_cast<const v8bf*>(bp + 16);   // K[16+8h..)
      v16bf bfrag = __builtin_shufflevector(blo, bhi,
                        0,1,2,3,4,5,6,7,8,9,10,11,12,13,14,15);
      acc = __builtin_amdgcn_wmma_f32_16x16x32_bf16(
                /*neg_a=*/false, afrag[j], /*neg_b=*/false, bfrag,
                /*c_mod=*/(short)0, acc, /*reuse_a=*/false, /*reuse_b=*/false);
    }

    // ---- fused epilogue: score[M] += sum_N acc[M,N] * F[M, it*16+N]
    // C/D layout: VGPR r -> M = r + 8h ; N = lane%16
    #pragma unroll
    for (int r = 0; r < 8; ++r) {
      float fv = __builtin_nontemporal_load(
          F + (size_t)(waveRow + r + 8 * h) * HID + it * 16 + m);
      score[r] += acc[r] * fv;
    }

    // ---- next chunk ready; one barrier per tile
    asm volatile("s_wait_asynccnt 0x0" ::: "memory");
    __syncthreads();
  }

  // ---- reduce each score over the 16 lanes of its half-wave
  #pragma unroll
  for (int r = 0; r < 8; ++r) {
    float vsum = score[r];
    vsum += __shfl_xor(vsum, 1, 32);
    vsum += __shfl_xor(vsum, 2, 32);
    vsum += __shfl_xor(vsum, 4, 32);
    vsum += __shfl_xor(vsum, 8, 32);
    score[r] = vsum;
  }
  if (m == 0) {
    #pragma unroll
    for (int r = 0; r < 8; ++r)
      __builtin_nontemporal_store(score[r], out + waveRow + r + 8 * h);
  }
}

extern "C" void kernel_launch(void* const* d_in, const int* in_sizes, int n_in,
                              void* d_out, int out_size, void* d_ws, size_t ws_size,
                              hipStream_t stream) {
  const float* F   = (const float*)d_in[0];  // features [B, 512]
  const float* S   = (const float*)d_in[1];  // summary  [B, 512]
  const float* W   = (const float*)d_in[2];  // weight   [512, 512]
  float*       out = (float*)d_out;          // scores   [B]
  bf16_t*      Wb  = (bf16_t*)d_ws;          // bf16 weight, 512 KB scratch

  const int wElems = in_sizes[2];            // 262144
  convert_w_bf16<<<wElems / (4 * 256), 256, 0, stream>>>(W, Wb);

  const int batch = in_sizes[0] / HID;
  const int grid  = batch / ROWS_PER_BLOCK;  // 131072/128 = 1024
  bilinear_score_wmma<<<grid, BLOCK, 0, stream>>>(F, S, Wb, out);
}
